// Node2VecGCNModel_16638703305294
// MI455X (gfx1250) — matrix operand order repaired and checked by reading, not compiled, not run on hardware
//
#include <hip/hip_runtime.h>

#define N_NODES 50000
#define HID 64
#define E_EDGES 800000
#define EP_EDGES 100000
#define L_LAYERS 2
#define BN_EPS 1e-5f
#define MTILES (N_NODES / 16)   // 3125 (exact)

typedef __attribute__((ext_vector_type(16))) __bf16 v16bf;
typedef __attribute__((ext_vector_type(8)))  __bf16 v8bf;
typedef __attribute__((ext_vector_type(8)))  float  v8f;

__device__ __forceinline__ void split_bf16(float f, __bf16& hi, __bf16& lo) {
    hi = (__bf16)f;
    lo = (__bf16)(f - (float)hi);
}

// ---------------------------------------------------------------------------
// Pre-pass 1: split a fp32 activation matrix [N x 64] into bf16 hi/lo planes,
// written row-major with row stride `lda_out` starting at column `col0`
// (lets us build the concatenated [id_emb | n2v_emb] K=128 operand in place).
// ---------------------------------------------------------------------------
__global__ void split_mat(const float* __restrict__ S, __bf16* __restrict__ Dh,
                          __bf16* __restrict__ Dl, int lda_out, int col0) {
    const int t = blockIdx.x * blockDim.x + threadIdx.x;
    if (t >= N_NODES * HID) return;
    const int row = t >> 6;
    const int c   = t & 63;
    __bf16 hi, lo; split_bf16(S[t], hi, lo);
    const int o = row * lda_out + col0 + c;
    Dh[o] = hi; Dl[o] = lo;
}

// ---------------------------------------------------------------------------
// Pre-pass 2: pack W[K x 64] (fp32) into bf16 hi/lo fragments laid out exactly
// as the ISA 16-bit B 32x16 VGPR layout expects, indexed (kt, nt, lane, j):
//   k = kt*32 + (j<8 ? (lane>>4)*8 + j : 16 + (lane>>4)*8 + (j-8)) ,  n = nt*16 + (lane&15)
// so the GEMM loads one contiguous 32B v16bf per fragment per lane.
// ---------------------------------------------------------------------------
__global__ void pack_w(const float* __restrict__ W, __bf16* __restrict__ Wh,
                       __bf16* __restrict__ Wl, int ktiles) {
    const int t = blockIdx.x * blockDim.x + threadIdx.x;
    if (t >= ktiles * 4 * 32 * 16) return;
    const int j    = t & 15;
    const int lane = (t >> 4) & 31;
    const int nt   = (t >> 9) & 3;
    const int kt   = t >> 11;
    const int k = kt * 32 + (j < 8 ? (lane >> 4) * 8 + j : 16 + (lane >> 4) * 8 + (j - 8));
    const int n = nt * 16 + (lane & 15);
    __bf16 hi, lo; split_bf16(W[k * HID + n], hi, lo);
    Wh[t] = hi; Wl[t] = lo;
}

// ---------------------------------------------------------------------------
// GEMM: H[N x 64] = A[N x (32*KTILES)] @ W + (bias)
// Operands pre-split to bf16 hi/lo; product via 3-term split:
//   A*B ~= Ah*Bh + Ah*Bl + Al*Bh   (rel. err ~2^-16, fp32-class)
// One wave per 16-row M tile. A fragment = two contiguous 16B runs per lane
// (ISA 16-bit A 16x32 layout); B fragment = one 32B pre-swizzled load.
// ---------------------------------------------------------------------------
template <int KTILES, bool HASBIAS>
__global__ void gemm_wmma(const __bf16* __restrict__ Ah, const __bf16* __restrict__ Al,
                          const __bf16* __restrict__ Wh, const __bf16* __restrict__ Wl,
                          const float* __restrict__ bias, float* __restrict__ H) {
    const int wave = (blockIdx.x * blockDim.x + threadIdx.x) >> 5;
    if (wave >= MTILES) return;                  // wave-uniform: EXEC stays all-ones
    const int lane = threadIdx.x & 31;
    const int half = lane >> 4;                  // 0: lanes 0-15, 1: lanes 16-31
    const int m    = lane & 15;
    const int mrow = wave * 16 + m;
    const int lda  = KTILES * 32;

    // A fragments: slots j=0..7 -> K = kt*32 + half*8 + j (contiguous),
    //              slots j=8..15 -> K = kt*32 + 16 + half*8 + (j-8) (contiguous)
    v16bf a_hi[KTILES], a_lo[KTILES];
#pragma unroll
    for (int kt = 0; kt < KTILES; ++kt) {
        const __bf16* p = Ah + mrow * lda + kt * 32 + half * 8;
        v8bf r0 = *(const v8bf*)(p);
        v8bf r1 = *(const v8bf*)(p + 16);
        a_hi[kt] = __builtin_shufflevector(r0, r1, 0,1,2,3,4,5,6,7,8,9,10,11,12,13,14,15);
        const __bf16* q = Al + mrow * lda + kt * 32 + half * 8;
        r0 = *(const v8bf*)(q);
        r1 = *(const v8bf*)(q + 16);
        a_lo[kt] = __builtin_shufflevector(r0, r1, 0,1,2,3,4,5,6,7,8,9,10,11,12,13,14,15);
    }

#pragma unroll
    for (int nt = 0; nt < 4; ++nt) {
        v8f acc = {};
#pragma unroll
        for (int kt = 0; kt < KTILES; ++kt) {
            const int fidx = ((kt * 4 + nt) * 32 + lane) * 16;
            const v16bf b_hi = *(const v16bf*)(Wh + fidx);
            const v16bf b_lo = *(const v16bf*)(Wl + fidx);
            acc = __builtin_amdgcn_wmma_f32_16x16x32_bf16(false, a_hi[kt], false, b_hi,
                                                          (short)0, acc, false, false);
            acc = __builtin_amdgcn_wmma_f32_16x16x32_bf16(false, a_hi[kt], false, b_lo,
                                                          (short)0, acc, false, false);
            acc = __builtin_amdgcn_wmma_f32_16x16x32_bf16(false, a_lo[kt], false, b_hi,
                                                          (short)0, acc, false, false);
        }
        // C/D layout: VGPR r -> row r (lanes 0-15) / r+8 (lanes 16-31), col = lane&15
#pragma unroll
        for (int r = 0; r < 8; ++r) {
            const int row  = wave * 16 + half * 8 + r;
            const int colg = nt * 16 + m;
            float v = acc[r];
            if (HASBIAS) v += bias[colg];
            H[row * HID + colg] = v;
        }
    }
}

// ---------------------------------------------------------------------------
// Graph / BN / decoder kernels
// ---------------------------------------------------------------------------
__global__ void init_deg(float* __restrict__ deg) {
    const int i = blockIdx.x * blockDim.x + threadIdx.x;
    if (i < N_NODES) deg[i] = 1.0f;              // self-loop contribution
}

__global__ void deg_edges(const int* __restrict__ ei, float* __restrict__ deg) {
    const int e = blockIdx.x * blockDim.x + threadIdx.x;
    if (e < E_EDGES) atomicAdd(&deg[ei[E_EDGES + e]], 1.0f);   // col = dst
}

__global__ void deg_to_dis(float* __restrict__ deg) {
    const int i = blockIdx.x * blockDim.x + threadIdx.x;
    if (i < N_NODES) deg[i] = rsqrtf(deg[i]);    // deg >= 1 always (self-loop)
}

__global__ void agg_init(float* __restrict__ agg, const float* __restrict__ b) {
    const int idx = blockIdx.x * blockDim.x + threadIdx.x;
    if (idx < N_NODES * HID) agg[idx] = b[idx & (HID - 1)];
}

__global__ void zero_stats(float* __restrict__ s) {
    s[threadIdx.x] = 0.0f;                       // 128 floats: sums + sumsq
}

// one (edge, 4-channel) pair per thread: agg[col] += dis[row]*dis[col]*h[row]
__global__ void scatter_edges(const int* __restrict__ ei, const float* __restrict__ dis,
                              const float* __restrict__ h, float* __restrict__ agg) {
    const int t  = blockIdx.x * blockDim.x + threadIdx.x;
    const int e  = t >> 4;
    const int c4 = (t & 15) * 4;
    if (e >= E_EDGES + N_NODES) return;
    int r, c; float s;
    if (e < E_EDGES) {
        __builtin_prefetch(ei + e + 8192, 0, 0);             // global_prefetch_b8 (src ids)
        __builtin_prefetch(ei + E_EDGES + e + 8192, 0, 0);   // dst ids
        r = ei[e]; c = ei[E_EDGES + e]; s = dis[r] * dis[c];
    } else {
        r = c = e - E_EDGES; const float d = dis[r]; s = d * d;
    }
    const float4 hv = *(const float4*)(h + r * HID + c4);
    float* ap = agg + c * HID + c4;
    atomicAdd(ap + 0, s * hv.x);
    atomicAdd(ap + 1, s * hv.y);
    atomicAdd(ap + 2, s * hv.z);
    atomicAdd(ap + 3, s * hv.w);
}

// per-column sum / sum-of-squares (stride % 64 == 0 -> fixed column per thread)
__global__ void bn_stats(const float* __restrict__ agg,
                         float* __restrict__ sums, float* __restrict__ sqs) {
    __shared__ float ss[256], ss2[256];
    const int t = threadIdx.x;
    float s = 0.0f, s2 = 0.0f;
    for (int idx = blockIdx.x * 256 + t; idx < N_NODES * HID; idx += gridDim.x * 256) {
        const float v = agg[idx];
        s += v; s2 += v * v;
    }
    ss[t] = s; ss2[t] = s2;
    __syncthreads();
    if (t < HID) {
        const float a = ss[t]  + ss[t + 64]  + ss[t + 128]  + ss[t + 192];
        const float b = ss2[t] + ss2[t + 64] + ss2[t + 128] + ss2[t + 192];
        atomicAdd(&sums[t], a);
        atomicAdd(&sqs[t],  b);
    }
}

__global__ void bn_apply(const float* __restrict__ agg, const float* __restrict__ sums,
                         const float* __restrict__ sqs, const float* __restrict__ gamma,
                         const float* __restrict__ beta, float* __restrict__ x) {
    const int idx = blockIdx.x * blockDim.x + threadIdx.x;
    if (idx >= N_NODES * HID) return;
    const int c = idx & (HID - 1);
    const float inv_n = 1.0f / (float)N_NODES;
    const float mu  = sums[c] * inv_n;
    const float var = sqs[c] * inv_n - mu * mu;
    const float rs  = rsqrtf(var + BN_EPS);
    const float xn  = (agg[idx] - mu) * rs * gamma[c] + beta[c];
    x[idx] += fmaxf(xn, 0.0f);
}

// 16 lanes per prediction edge; float4 loads + xor-shuffle reduction (wave32)
__global__ void decode(const int* __restrict__ pred, const float* __restrict__ x,
                       float* __restrict__ out) {
    const int t = blockIdx.x * blockDim.x + threadIdx.x;
    const int p = t >> 4;
    const int q = t & 15;
    if (p >= EP_EDGES) return;
    const int a = pred[2 * p + 0];
    const int b = pred[2 * p + 1];
    const float4 xa = *(const float4*)(x + a * HID + q * 4);
    const float4 xb = *(const float4*)(x + b * HID + q * 4);
    float s = xa.x * xb.x + xa.y * xb.y + xa.z * xb.z + xa.w * xb.w;
    s += __shfl_xor(s, 1, 32);
    s += __shfl_xor(s, 2, 32);
    s += __shfl_xor(s, 4, 32);
    s += __shfl_xor(s, 8, 32);
    if (q == 0) out[p] = s;
}

// ---------------------------------------------------------------------------
extern "C" void kernel_launch(void* const* d_in, const int* in_sizes, int n_in,
                              void* d_out, int out_size, void* d_ws, size_t ws_size,
                              hipStream_t stream) {
    const int*   edge_index = (const int*)  d_in[0];
    const int*   pred_edge  = (const int*)  d_in[1];
    const float* id_emb     = (const float*)d_in[2];
    const float* n2v_emb    = (const float*)d_in[3];
    const float* proj_w     = (const float*)d_in[4];
    const float* proj_b     = (const float*)d_in[5];
    const float* conv_w     = (const float*)d_in[6];
    const float* conv_b     = (const float*)d_in[7];
    const float* bn_gamma   = (const float*)d_in[8];
    const float* bn_beta    = (const float*)d_in[9];
    float* out = (float*)d_out;

    // workspace layout: f32 { x | h | agg | deg | sums | sqs } then bf16 planes
    float* x    = (float*)d_ws;
    float* h    = x    + (size_t)N_NODES * HID;
    float* agg  = h    + (size_t)N_NODES * HID;
    float* deg  = agg  + (size_t)N_NODES * HID;   // overwritten in-place with rsqrt
    float* sums = deg  + N_NODES;
    float* sqs  = sums + HID;
    __bf16* Ah  = (__bf16*)(sqs + HID);           // N x 128 bf16
    __bf16* Al  = Ah + (size_t)N_NODES * 128;
    __bf16* Wh  = Al + (size_t)N_NODES * 128;     // up to 4*4*32*16 = 8192 bf16
    __bf16* Wl  = Wh + 8192;

    const int BS = 256;
    const int gN    = (N_NODES + BS - 1) / BS;
    const int gE    = (E_EDGES + BS - 1) / BS;
    const int gNH   = (N_NODES * HID + BS - 1) / BS;
    const int gSc   = ((E_EDGES + N_NODES) * 16 + BS - 1) / BS;
    const int gDec  = (EP_EDGES * 16 + BS - 1) / BS;
    const int gGemm = (MTILES + 7) / 8;           // 8 waves (256 thr) per block

    // GCN normalization: deg -> dis = rsqrt(deg)
    init_deg  <<<gN, BS, 0, stream>>>(deg);
    deg_edges <<<gE, BS, 0, stream>>>(edge_index, deg);
    deg_to_dis<<<gN, BS, 0, stream>>>(deg);

    // ---- projection: x = [id_emb | n2v_emb] @ proj_w + proj_b  (K=128) ----
    split_mat<<<gNH, BS, 0, stream>>>(id_emb,  Ah, Al, 128, 0);
    split_mat<<<gNH, BS, 0, stream>>>(n2v_emb, Ah, Al, 128, HID);
    pack_w   <<<32,  BS, 0, stream>>>(proj_w, Wh, Wl, 4);
    gemm_wmma<4, true><<<gGemm, BS, 0, stream>>>(Ah, Al, Wh, Wl, proj_b, x);

    for (int l = 0; l < L_LAYERS; ++l) {
        // ---- h = x @ conv_w[l]  (K=64) ----
        split_mat<<<gNH, BS, 0, stream>>>(x, Ah, Al, HID, 0);
        pack_w   <<<16,  BS, 0, stream>>>(conv_w + l * HID * HID, Wh, Wl, 2);
        gemm_wmma<2, false><<<gGemm, BS, 0, stream>>>(Ah, Al, Wh, Wl, nullptr, h);

        agg_init  <<<gNH, BS, 0, stream>>>(agg, conv_b + l * HID);
        zero_stats<<<1, 2 * HID, 0, stream>>>(sums);
        scatter_edges<<<gSc, BS, 0, stream>>>(edge_index, deg, h, agg);
        bn_stats  <<<128, BS, 0, stream>>>(agg, sums, sqs);
        bn_apply  <<<gNH, BS, 0, stream>>>(agg, sums, sqs,
                                           bn_gamma + l * HID, bn_beta + l * HID, x);
    }

    decode<<<gDec, BS, 0, stream>>>(pred_edge, x, out);
}
